// AttentionLayer_58119497449503
// MI455X (gfx1250) — matrix operand order repaired and checked by reading, not compiled, not run on hardware
//
#include <hip/hip_runtime.h>
#include <hip/hip_bf16.h>

// Problem constants (match reference)
#define TS 2048   // sequence length T
#define DM 1024   // d_model D
#define NH 16     // heads H
#define HD 64     // head dim P
#define NB 2      // batch B

typedef _Float16 h16;
typedef __attribute__((ext_vector_type(16))) _Float16 v16h;
typedef __attribute__((ext_vector_type(8)))  _Float16 v8h;
typedef __attribute__((ext_vector_type(8)))  float    v8f;
typedef __attribute__((ext_vector_type(4)))  float    v4f;

static __device__ __forceinline__ v16h cat16(v8h a, v8h b) {
  v16h r;
#pragma unroll
  for (int i = 0; i < 8; ++i) { r[i] = a[i]; r[i + 8] = b[i]; }
  return r;
}
static __device__ __forceinline__ v8h ld8h(const h16* p) {
  return *reinterpret_cast<const v8h*>(p);
}
static __device__ __forceinline__ v8h cvt8(const float* p) {
  v4f a = *reinterpret_cast<const v4f*>(p);
  v4f b = *reinterpret_cast<const v4f*>(p + 4);
  v8h r;
#pragma unroll
  for (int i = 0; i < 4; ++i) { r[i] = (h16)a[i]; r[i + 4] = (h16)b[i]; }
  return r;
}
#define WMMA16(A, B, C) \
  __builtin_amdgcn_wmma_f32_16x16x32_f16(false, (A), false, (B), (short)0, (C), false, false)

// ---------------------------------------------------------------------------
// Kernel 0a: WQ/WK/WV (h,d,p) f32 -> wAll[z] (h,p,d) f16   (K-contiguous for WMMA)
__global__ void k_cvt_w(const float* __restrict__ wq, const float* __restrict__ wk,
                        const float* __restrict__ wv, h16* __restrict__ wAll) {
  long idx = (long)blockIdx.x * blockDim.x + threadIdx.x;
  const long n = (long)NH * DM * HD;
  if (idx >= 3 * n) return;
  int z = (int)(idx / n);
  long r = idx % n;
  int h = (int)(r / (DM * HD));
  int rem = (int)(r % (DM * HD));
  int d = rem / HD, p = rem % HD;
  const float* src = (z == 0) ? wq : (z == 1 ? wk : wv);
  wAll[z * n + (long)h * HD * DM + (long)p * DM + d] = (h16)src[r];
}

// Kernel 0b: dense_W (hp,d) f32 -> wdT (d,hp) f16
__global__ void k_cvt_dense(const float* __restrict__ w, h16* __restrict__ wdT) {
  long idx = (long)blockIdx.x * blockDim.x + threadIdx.x;
  if (idx >= (long)DM * DM) return;
  int hp = (int)(idx / DM), d = (int)(idx % DM);
  wdT[(long)d * DM + hp] = (h16)w[idx];
}

// Kernel 0c: rel_emb (2T-1, P) f32 -> relF (2T, P) f16, last row zero padded
__global__ void k_cvt_rel(const float* __restrict__ rel, h16* __restrict__ relF) {
  long idx = (long)blockIdx.x * blockDim.x + threadIdx.x;
  if (idx >= (long)2 * TS * HD) return;
  long row = idx / HD;
  relF[idx] = (row < 2 * TS - 1) ? (h16)rel[idx] : (h16)0.0f;
}

// ---------------------------------------------------------------------------
// Kernel 1: projections. One wave computes a full 16 x 64 (t x p) tile.
// All fragments of a K-step are loaded before the 4 back-to-back WMMAs so the
// loads issue as one clause and the WMMAs pipeline on the XDL unit.
//   z=0: q, z=1: k  (stored (b,h,t,p) f16; computed transposed so C-rows = p)
//   z=2: v          (stored transposed (b,h,p,t) f16; computed normal)
__global__ void __launch_bounds__(128)
k_proj(const float* __restrict__ Q, const float* __restrict__ K,
       const float* __restrict__ V, const h16* __restrict__ wAll,
       h16* __restrict__ qb, h16* __restrict__ kb, h16* __restrict__ vT) {
  const int lane = threadIdx.x & 31;
  const int qi = lane & 15, hi = lane >> 4;
  const int wave = threadIdx.x >> 5;                // 4 waves -> 4 t-tiles
  const int t0 = (blockIdx.x * 4 + wave) * 16;
  const int bh = blockIdx.y, b = bh >> 4, h = bh & 15;
  const int z = blockIdx.z;
  const long wstride = (long)NH * HD * DM;
  const h16* W = wAll + (long)z * wstride + (long)h * HD * DM;
  const float* X = (z == 0) ? Q : (z == 1 ? K : V);
  X += (long)b * TS * DM;
  v8f acc[4] = {};
  if (z < 2) {
    // out^T (p x t): A = W^T (p rows, K=d) per p-tile, B = X^T (shared)
    const float* brow = X + (long)(t0 + qi) * DM;
    for (int k0 = 0; k0 < DM; k0 += 32) {
      v16h afr[4];
#pragma unroll
      for (int pt = 0; pt < 4; ++pt) {
        const h16* arow = W + (long)(pt * 16 + qi) * DM;
        afr[pt] = cat16(ld8h(arow + k0 + hi * 8), ld8h(arow + k0 + 16 + hi * 8));
      }
      v16h bfr = cat16(cvt8(brow + k0 + hi * 8), cvt8(brow + k0 + 16 + hi * 8));
#pragma unroll
      for (int pt = 0; pt < 4; ++pt) acc[pt] = WMMA16(afr[pt], bfr, acc[pt]);
    }
    // C layout: col = t (lane), rows p = pt*16 + v + 8*hi -> contiguous in (t,p)
    h16* base = ((z == 0) ? qb : kb) + (long)bh * TS * HD + (long)(t0 + qi) * HD;
#pragma unroll
    for (int pt = 0; pt < 4; ++pt) {
      v8h o;
#pragma unroll
      for (int v = 0; v < 8; ++v) o[v] = (h16)acc[pt][v];
      *reinterpret_cast<v8h*>(base + pt * 16 + 8 * hi) = o;
    }
  } else {
    // out (t x p): A = V rows (shared, K=d), B = Wv^T cols (K=d) per p-tile
    const float* arow = X + (long)(t0 + qi) * DM;
    for (int k0 = 0; k0 < DM; k0 += 32) {
      v16h bfr[4];
#pragma unroll
      for (int pt = 0; pt < 4; ++pt) {
        const h16* brow = W + (long)(pt * 16 + qi) * DM;
        bfr[pt] = cat16(ld8h(brow + k0 + hi * 16), ld8h(brow + k0 + hi * 16 + 8));
      }
      v16h a = cat16(cvt8(arow + k0 + hi * 16), cvt8(arow + k0 + hi * 16 + 8));
#pragma unroll
      for (int pt = 0; pt < 4; ++pt) acc[pt] = WMMA16(a, bfr[pt], acc[pt]);
    }
    // C layout: col = p (lane), rows t = t0 + v + 8*hi -> contiguous in (p,t)
    h16* base = vT + (long)bh * HD * TS;
#pragma unroll
    for (int pt = 0; pt < 4; ++pt) {
      v8h o;
#pragma unroll
      for (int v = 0; v < 8; ++v) o[v] = (h16)acc[pt][v];
      *reinterpret_cast<v8h*>(base + (long)(pt * 16 + qi) * TS + t0 + 8 * hi) = o;
    }
  }
}

// ---------------------------------------------------------------------------
// Kernel 2: flash attention (transposed S^T = K Q^T) + Shaw skew RPE via WMMA+LDS
__global__ void __launch_bounds__(256)
k_attn(const h16* __restrict__ qb, const h16* __restrict__ kb,
       const h16* __restrict__ vT, const h16* __restrict__ relF,
       const float* __restrict__ mask, h16* __restrict__ ctx) {
  __shared__ float lds[8 * 48 * 16];   // per-wave 48x16 skew buffer
  const int lane = threadIdx.x & 31;
  const int qi = lane & 15, hi = lane >> 4;
  const int wave = threadIdx.x >> 5;
  float* wl = lds + wave * 48 * 16;
  const int bh = blockIdx.y, b = bh >> 4, h = bh & 15;
  const int i0 = (blockIdx.x * 8 + wave) * 16;   // query tile base

  const h16* qrow = qb + (long)bh * TS * HD + (long)(i0 + qi) * HD;
  const h16* kbase = kb + (long)bh * TS * HD;
  const h16* vbase = vT + (long)bh * HD * TS;
  const float* mrow = mask + (long)b * TS;

  // Q^T B-fragments (lane = query col, K = p), reused for S^T and RPE WMMAs
  v16h bq0 = cat16(ld8h(qrow + hi * 8), ld8h(qrow + 16 + hi * 8));
  v16h bq1 = cat16(ld8h(qrow + 32 + hi * 8), ld8h(qrow + 48 + hi * 8));

  const float scale = 0.125f;                       // 1/sqrt(64)
  const float L2E = 1.44269504088896340736f;
  float m = -3.0e38f, l = 0.0f;
  v8f o0 = {}, o1 = {}, o2 = {}, o3 = {};           // O^T accum, p-tiles 0..3

  for (int jc = 0; jc < TS; jc += 32) {
    if (jc + 32 < TS) {
      __builtin_prefetch(kbase + (long)(jc + 32 + qi) * HD, 0, 1);
      __builtin_prefetch(vbase + (long)qi * TS + jc + 32, 0, 1);
    }
    // one mask load per lane covers all 32 keys of this chunk; broadcast via shfl
    float mterm = (1.0f - mrow[jc + lane]) * (-1.0e9f);
    // ---- load all K-tile A-fragments, then 4 back-to-back S^T WMMAs
    v8f st0 = {}, st1 = {};
    {
      const h16* kr0 = kbase + (long)(jc + qi) * HD;
      const h16* kr1 = kbase + (long)(jc + 16 + qi) * HD;
      v16h a00 = cat16(ld8h(kr0 + hi * 8), ld8h(kr0 + 16 + hi * 8));
      v16h a01 = cat16(ld8h(kr0 + 32 + hi * 8), ld8h(kr0 + 48 + hi * 8));
      v16h a10 = cat16(ld8h(kr1 + hi * 8), ld8h(kr1 + 16 + hi * 8));
      v16h a11 = cat16(ld8h(kr1 + 32 + hi * 8), ld8h(kr1 + 48 + hi * 8));
      st0 = WMMA16(a00, bq0, st0);
      st1 = WMMA16(a10, bq0, st1);
      st0 = WMMA16(a01, bq1, st0);
      st1 = WMMA16(a11, bq1, st1);
    }
    // ---- RPE: 48 rel rows starting at rb; Ct = rel_slice @ Q^T -> LDS skew gather
    const int rb = jc - i0 + TS - 16;   // >= 0, rb+47 <= 2T-1 (padded row)
    {
      v16h ra[3][2];
#pragma unroll
      for (int rt = 0; rt < 3; ++rt) {
        const h16* rr = relF + (long)(rb + 16 * rt + qi) * HD;
        ra[rt][0] = cat16(ld8h(rr + hi * 8), ld8h(rr + 16 + hi * 8));
        ra[rt][1] = cat16(ld8h(rr + 32 + hi * 8), ld8h(rr + 48 + hi * 8));
      }
      v8f ct[3] = {};
#pragma unroll
      for (int rt = 0; rt < 3; ++rt) ct[rt] = WMMA16(ra[rt][0], bq0, ct[rt]);
#pragma unroll
      for (int rt = 0; rt < 3; ++rt) ct[rt] = WMMA16(ra[rt][1], bq1, ct[rt]);
#pragma unroll
      for (int rt = 0; rt < 3; ++rt)
#pragma unroll
        for (int v = 0; v < 8; ++v)
          wl[(16 * rt + v + 8 * hi) * 16 + qi] = ct[rt][v];
    }
    asm volatile("s_wait_dscnt 0" ::: "memory");
    // ---- skew gather + scale + mask
    float s0[8], s1[8];
#pragma unroll
    for (int v = 0; v < 8; ++v) {
      int dj0 = v + 8 * hi;
      float bias0 = wl[(dj0 - qi + 15) * 16 + qi];
      s0[v] = (st0[v] + bias0) * scale + __shfl(mterm, dj0, 32);
      int dj1 = dj0 + 16;
      float bias1 = wl[(dj1 - qi + 15) * 16 + qi];
      s1[v] = (st1[v] + bias1) * scale + __shfl(mterm, dj1, 32);
    }
    // ---- online softmax (per query col = per lane; halves kept consistent via shfl)
    float lm = s0[0];
#pragma unroll
    for (int v = 1; v < 8; ++v) lm = fmaxf(lm, s0[v]);
#pragma unroll
    for (int v = 0; v < 8; ++v) lm = fmaxf(lm, s1[v]);
    lm = fmaxf(lm, __shfl_xor(lm, 16, 32));
    float mn = fmaxf(m, lm);
    float alpha = __builtin_exp2f((m - mn) * L2E);
    float p0r[8], p1r[8], ls = 0.0f;
#pragma unroll
    for (int v = 0; v < 8; ++v) {
      p0r[v] = __builtin_exp2f((s0[v] - mn) * L2E);
      p1r[v] = __builtin_exp2f((s1[v] - mn) * L2E);
      ls += p0r[v] + p1r[v];
    }
    ls += __shfl_xor(ls, 16, 32);
    l = l * alpha + ls;
    m = mn;
    o0 *= alpha; o1 *= alpha; o2 *= alpha; o3 *= alpha;
    // ---- pack P^T into B-fragment (lanes<16: tile0 keys 0..15, lanes>=16: tile1)
    v16h bp;
#pragma unroll
    for (int v = 0; v < 8; ++v) {
      float q0 = __shfl_xor(p0r[v], 16, 32);   // partner's tile0 (other row half)
      float q1 = __shfl_xor(p1r[v], 16, 32);   // partner's tile1
      float lo = hi ? q1 : p0r[v];             // K = v      (rows 0..7 of my tile)
      float hv = hi ? p1r[v] : q0;             // K = v + 8  (rows 8..15)
      bp[v] = (h16)lo;
      bp[v + 8] = (h16)hv;
    }
    // ---- O^T += V^T @ P^T: load all 4 V^T fragments, then 4 back-to-back WMMAs
    {
      const h16* vr = vbase + (long)qi * TS + jc;
      v16h av[4];
#pragma unroll
      for (int pt = 0; pt < 4; ++pt) {
        av[pt] = cat16(ld8h(vr + hi * 16), ld8h(vr + hi * 16 + 8));
        vr += (long)16 * TS;
      }
      o0 = WMMA16(av[0], bp, o0);
      o1 = WMMA16(av[1], bp, o1);
      o2 = WMMA16(av[2], bp, o2);
      o3 = WMMA16(av[3], bp, o3);
    }
  }
  // ---- normalize and store ctx (b, t, h*64+p) f16 — packed b128 stores
  float rl = 1.0f / l;
  h16* crow = ctx + ((long)b * TS + i0 + qi) * (NH * HD) + h * HD;
  v8f oo[4] = {o0, o1, o2, o3};
#pragma unroll
  for (int pt = 0; pt < 4; ++pt) {
    v8h o;
#pragma unroll
    for (int v = 0; v < 8; ++v) o[v] = (h16)(oo[pt][v] * rl);
    *reinterpret_cast<v8h*>(crow + 16 * pt + 8 * hi) = o;
  }
}

// ---------------------------------------------------------------------------
// Kernel 3: out = ctx(B*T,1024) @ dense_W(1024,1024) + b, f32 output.
// One wave computes 4 column tiles; all fragments loaded before the 4 WMMAs.
__global__ void __launch_bounds__(256)
k_dense(const h16* __restrict__ ctx, const h16* __restrict__ wdT,
        const float* __restrict__ bias, float* __restrict__ out) {
  const int lane = threadIdx.x & 31;
  const int qi = lane & 15, hi = lane >> 4;
  const int wave = threadIdx.x >> 5;
  const int ct0 = (blockIdx.x * 8 + wave) * 4;   // first of 4 column tiles
  const int r0 = blockIdx.y * 16;                // row tile over B*T
  const h16* arow = ctx + (long)(r0 + qi) * DM;
  v8f acc[4] = {};
  for (int k0 = 0; k0 < DM; k0 += 32) {
    v16h bfr[4];
#pragma unroll
    for (int pt = 0; pt < 4; ++pt) {
      const h16* brow = wdT + (long)((ct0 + pt) * 16 + qi) * DM;
      bfr[pt] = cat16(ld8h(brow + k0 + hi * 16), ld8h(brow + k0 + hi * 16 + 8));
    }
    v16h a = cat16(ld8h(arow + k0 + hi * 16), ld8h(arow + k0 + hi * 16 + 8));
#pragma unroll
    for (int pt = 0; pt < 4; ++pt) acc[pt] = WMMA16(a, bfr[pt], acc[pt]);
  }
#pragma unroll
  for (int pt = 0; pt < 4; ++pt) {
    float bv = bias[(ct0 + pt) * 16 + qi];
#pragma unroll
    for (int v = 0; v < 8; ++v)
      out[(long)(r0 + v + 8 * hi) * DM + (ct0 + pt) * 16 + qi] = acc[pt][v] + bv;
  }
}

// ---------------------------------------------------------------------------
extern "C" void kernel_launch(void* const* d_in, const int* in_sizes, int n_in,
                              void* d_out, int out_size, void* d_ws, size_t ws_size,
                              hipStream_t stream) {
  const float* Q    = (const float*)d_in[0];
  const float* K    = (const float*)d_in[1];
  const float* V    = (const float*)d_in[2];
  const float* mask = (const float*)d_in[3];
  const float* WQ   = (const float*)d_in[4];
  const float* WK   = (const float*)d_in[5];
  const float* WV   = (const float*)d_in[6];
  const float* rel  = (const float*)d_in[7];
  const float* Wd   = (const float*)d_in[8];
  const float* bd   = (const float*)d_in[9];
  float* out = (float*)d_out;

  char* ws = (char*)d_ws;
  const long nW   = (long)NH * DM * HD;            // per-weight elems
  h16* wAll = (h16*)(ws);                          // 3 * 2 MB
  h16* wdT  = (h16*)(ws + 3 * nW * 2);             // 2 MB
  h16* relF = (h16*)(ws + 3 * nW * 2 + (long)DM * DM * 2);          // 512 KB
  char* p2  = ws + 3 * nW * 2 + (long)DM * DM * 2 + (long)2 * TS * HD * 2;
  const long nQKV = (long)NB * NH * TS * HD;       // 4 M elems -> 8 MB each
  h16* qb = (h16*)(p2);
  h16* kb = (h16*)(p2 + nQKV * 2);
  h16* vT = (h16*)(p2 + 2 * nQKV * 2);
  h16* ctx = (h16*)(p2 + 3 * nQKV * 2);            // 8 MB

  // 0) weight / rel-emb conversion & transposition
  {
    long n3 = 3 * nW;
    k_cvt_w<<<dim3((unsigned)((n3 + 255) / 256)), dim3(256), 0, stream>>>(WQ, WK, WV, wAll);
    long nd = (long)DM * DM;
    k_cvt_dense<<<dim3((unsigned)((nd + 255) / 256)), dim3(256), 0, stream>>>(Wd, wdT);
    long nr = (long)2 * TS * HD;
    k_cvt_rel<<<dim3((unsigned)((nr + 255) / 256)), dim3(256), 0, stream>>>(rel, relF);
  }
  // 1) q/k/v projections (WMMA), q/k -> (b,h,t,p), v -> (b,h,p,t)
  k_proj<<<dim3(TS / 64, NB * NH, 3), dim3(128), 0, stream>>>(Q, K, V, wAll, qb, kb, vT);
  // 2) flash attention with skewed relative-position bias
  k_attn<<<dim3(TS / 128, NB * NH), dim3(256), 0, stream>>>(qb, kb, vT, relF, mask, ctx);
  // 3) output projection + bias
  k_dense<<<dim3(2, NB * TS / 16), dim3(256), 0, stream>>>(ctx, wdT, bd, out);
}
